// Attention_37512244363591
// MI455X (gfx1250) — matrix operand order repaired and checked by reading, not compiled, not run on hardware
//
#include <hip/hip_runtime.h>

typedef __bf16 bf16;
typedef __attribute__((ext_vector_type(16))) __bf16 v16bf;
typedef __attribute__((ext_vector_type(8)))  float  v8f;

#define DEV static __device__ __forceinline__

constexpr int B_ = 2, SQ = 2048, HD = 1024, NH = 16, DH = 64;
constexpr int QKV_COLS = 3 * NH * DH;  // 3072

DEV v16bf load_v16bf(const bf16* p) {
  union { uint4 q[2]; v16bf v; } u;
  u.q[0] = *reinterpret_cast<const uint4*>(p);
  u.q[1] = *reinterpret_cast<const uint4*>(p + 8);
  return u.v;
}

// A-fragment (16x32 M x K) from row-major tile, leading dim ld (elements).
// Lane m = lane&15; half = lane>>4; elem i -> K = half*16+i.
DEV v16bf load_a_frag(const bf16* tile, int ld, int lane) {
  int m = lane & 15, h = lane >> 4;
  const bf16* r = tile + m * ld;
  union { uint4 q[2]; v16bf v; } u;
  u.q[0] = *reinterpret_cast<const uint4*>(r + h * 16);
  u.q[1] = *reinterpret_cast<const uint4*>(r + h * 16 + 8);
  return u.v;
}

// B-fragment (32x16 K x N) from K-contiguous ([n][k]) tile, leading dim ld.
DEV v16bf load_b_fragT(const bf16* tileT, int ld, int lane) {
  int n = lane & 15, h = lane >> 4;
  return load_v16bf(tileT + n * ld + h * 16);
}

DEV v8f wmma_bf16(v16bf a, v16bf b, v8f c) {
  return __builtin_amdgcn_wmma_f32_16x16x32_bf16(false, a, false, b, (short)0, c,
                                                 false, false);
}

DEV unsigned pack2(float a, float b) {  // two f32 -> packed bf16x2
  union { bf16 h[2]; unsigned u; } p;
  p.h[0] = (bf16)a; p.h[1] = (bf16)b;
  return p.u;
}

// CDNA5 async copy: global -> LDS, 16B per lane, tracked by ASYNCcnt.
DEV void async_b128(unsigned lds_off, const bf16* g) {
  asm volatile("global_load_async_to_lds_b128 %0, %1, off"
               :: "v"(lds_off), "v"((unsigned long long)(size_t)g)
               : "memory");
}
DEV void wait_async0() { asm volatile("s_wait_asynccnt 0x0" ::: "memory"); }
DEV unsigned lds_off(const void* p) { return (unsigned)(size_t)p; }

// ---------------------------------------------------------------------------
// Kernel 1: per-head inclusive cumsum of rpe_bias (wave-parallel scan).
// ---------------------------------------------------------------------------
__global__ void rpe_cumsum_kernel(const float* __restrict__ rpe,
                                  float* __restrict__ cum) {
  int n = blockIdx.x, lane = threadIdx.x;
  float carry = 0.f;
  for (int s0 = 0; s0 < SQ; s0 += 32) {
    float v = rpe[n * SQ + s0 + lane];
#pragma unroll
    for (int off = 1; off < 32; off <<= 1) {
      float t = __shfl_up(v, off, 32);
      if (lane >= off) v += t;
    }
    float out = v + carry;
    cum[n * SQ + s0 + lane] = out;
    carry = __shfl(out, 31, 32);
  }
}

// ---------------------------------------------------------------------------
// Kernel 2: QKV projection. X[4096,1024] * W[1024,3072] -> scatter bf16
// Q,K as [b,n,s,d]; V transposed as [b,n,d,s]. Q pre-scaled by D^-0.5.
// Double-buffered LDS: stage tile i+1 while WMMAs consume tile i.
// ---------------------------------------------------------------------------
DEV void gemm_stage_f32(const float* a, int lda, const float* b, int ldb,
                        bf16* As, bf16* Bs, int tid) {
  {  // A tile 128x32: f32 -> packed bf16x2
    int row = tid >> 1, kk = (tid & 1) * 16;
    const float* src = a + (size_t)row * lda + kk;
    unsigned* dst = reinterpret_cast<unsigned*>(As + row * 48 + kk);
#pragma unroll
    for (int j = 0; j < 8; ++j) dst[j] = pack2(src[2 * j], src[2 * j + 1]);
  }
  {  // B tile 32x128 -> transposed [n][k]: k-pair x 8-n strip, packed b32
    int kp = tid >> 4;
    int n0 = (tid & 15) * 8;
    const float* s0 = b + (size_t)(2 * kp) * ldb + n0;
    const float* s1 = s0 + ldb;
#pragma unroll
    for (int j = 0; j < 8; ++j)
      *reinterpret_cast<unsigned*>(Bs + (n0 + j) * 48 + 2 * kp) =
          pack2(s0[j], s1[j]);
  }
}

__global__ __launch_bounds__(256) void qkv_proj_kernel(
    const float* __restrict__ x, const float* __restrict__ w,
    bf16* __restrict__ q, bf16* __restrict__ k, bf16* __restrict__ vt) {
  __shared__ alignas(16) bf16 As[2][128 * 48];
  __shared__ alignas(16) bf16 Bs[2][128 * 48];
  int tid = threadIdx.x, wv = tid >> 5, lane = tid & 31;
  int half = lane >> 4, l15 = lane & 15;
  int r0 = blockIdx.y * 128;
  int c0 = blockIdx.x * 128;
  int wm = (wv >> 2) * 64, wn = (wv & 3) * 32;
  v8f acc[4][2] = {};
  gemm_stage_f32(x + (size_t)r0 * HD, HD, w + c0, QKV_COLS, As[0], Bs[0], tid);
  int buf = 0;
  for (int k0 = 0; k0 < HD; k0 += 32) {
    __syncthreads();  // stage of `buf` complete everywhere; prior reads done
    if (k0 + 32 < HD)
      gemm_stage_f32(x + (size_t)r0 * HD + k0 + 32, HD,
                     w + (size_t)(k0 + 32) * QKV_COLS + c0, QKV_COLS,
                     As[buf ^ 1], Bs[buf ^ 1], tid);
    v16bf af[4], bfr[2];
#pragma unroll
    for (int mi = 0; mi < 4; ++mi)
      af[mi] = load_a_frag(As[buf] + (wm + mi * 16) * 48, 48, lane);
#pragma unroll
    for (int ni = 0; ni < 2; ++ni)
      bfr[ni] = load_b_fragT(Bs[buf] + (wn + ni * 16) * 48, 48, lane);
#pragma unroll
    for (int mi = 0; mi < 4; ++mi)
#pragma unroll
      for (int ni = 0; ni < 2; ++ni)
        acc[mi][ni] = wmma_bf16(af[mi], bfr[ni], acc[mi][ni]);
    buf ^= 1;
  }
  // scatter epilogue: tile-uniform base pointers, constant-offset stores
#pragma unroll
  for (int mi = 0; mi < 4; ++mi)
#pragma unroll
    for (int ni = 0; ni < 2; ++ni) {
      int c = c0 + wn + ni * 16 + l15;
      int m3 = c >> 10;            // 0=q,1=k,2=v (wave-uniform per tile)
      int nd = c & 1023;
      int n = nd >> 6, d = nd & 63;
      int rbase = r0 + wm + mi * 16;            // 16-aligned
      int b = rbase >> 11, s0 = (rbase & (SQ - 1)) + 8 * half;
      size_t bn = (size_t)b * NH + n;
      if (m3 == 0) {
        bf16* p = q + (bn * SQ + s0) * DH + d;
#pragma unroll
        for (int e = 0; e < 8; ++e) p[e * DH] = (bf16)(acc[mi][ni][e] * 0.125f);
      } else if (m3 == 1) {
        bf16* p = k + (bn * SQ + s0) * DH + d;
#pragma unroll
        for (int e = 0; e < 8; ++e) p[e * DH] = (bf16)acc[mi][ni][e];
      } else {
        bf16* p = vt + (bn * DH + d) * SQ + s0;
#pragma unroll
        for (int e = 0; e < 8; ++e) p[e] = (bf16)acc[mi][ni][e];
      }
    }
}

// ---------------------------------------------------------------------------
// Kernel 3: causal flash attention with cumulative RPE bias.
// Block = 128 rows of one (b,n). K/V tiles double-buffered via
// global_load_async_to_lds_b128: DMA of tile j+1 overlaps compute of tile j.
// ---------------------------------------------------------------------------
__global__ __launch_bounds__(256) void attn_kernel(
    const bf16* __restrict__ q, const bf16* __restrict__ k,
    const bf16* __restrict__ vt, const float* __restrict__ cum,
    bf16* __restrict__ mix) {
  __shared__ alignas(16) float cum_s[SQ];          // 8 KB
  __shared__ alignas(16) bf16 Ks[2][32 * 64];      // K tiles [j][d], 8 KB
  __shared__ alignas(16) bf16 Vs[2][64 * 32];      // V tiles [d][j], 8 KB
  __shared__ alignas(16) bf16 p_s[8][16 * 48];     // per-wave P re-swizzle
  int tid = threadIdx.x, wv = tid >> 5, lane = tid & 31;
  int half = lane >> 4, l15 = lane & 15;
  int bn = blockIdx.y;
  int n = bn & (NH - 1);
  int base = blockIdx.x * 128;
  int i0 = base + wv * 16;

  for (int t = tid; t < SQ; t += 256) cum_s[t] = cum[n * SQ + t];

  const bf16* qt = q + ((size_t)bn * SQ + i0) * DH;
  v16bf aq0 = load_a_frag(qt, DH, lane);        // d = 0..31
  v16bf aq1 = load_a_frag(qt + 32, DH, lane);   // d = 32..63

  v8f o[4] = {};
  float m_i[8], l_i[8];
#pragma unroll
  for (int e = 0; e < 8; ++e) { m_i[e] = -3.0e38f; l_i[e] = 0.f; }

  const bf16* kb = k + (size_t)bn * SQ * DH;
  const bf16* vb = vt + (size_t)bn * DH * SQ;
  bf16* pw = &p_s[wv][0];

  // per-thread async staging slots (one K b128 + one V b128 per thread)
  int kjr = tid >> 3, kch = tid & 7;     // K: 32 rows x 8 chunks of 8 elem
  int vdr = tid >> 2, vch = tid & 3;     // V: 64 rows x 4 chunks of 8 elem
  unsigned koff[2] = { lds_off(Ks[0] + kjr * 64 + kch * 8),
                       lds_off(Ks[1] + kjr * 64 + kch * 8) };
  unsigned voff[2] = { lds_off(Vs[0] + vdr * 32 + vch * 8),
                       lds_off(Vs[1] + vdr * 32 + vch * 8) };
  const bf16* kg = kb + (size_t)kjr * DH + kch * 8;
  const bf16* vg = vb + (size_t)vdr * SQ + vch * 8;

  int jmax = base + 128;                 // block-uniform causal bound
  async_b128(koff[0], kg);               // prologue: stage tile 0
  async_b128(voff[0], vg);
  int buf = 0;
  for (int j0 = 0; j0 < jmax; j0 += 32) {
    wait_async0();                       // own copies into `buf` done
    __syncthreads();                     // everyone's copies done / reads done
    if (j0 + 32 < jmax) {                // prefetch tile j+1 into other buffer
      async_b128(koff[buf ^ 1], kg + (size_t)(j0 + 32) * DH);
      async_b128(voff[buf ^ 1], vg + (j0 + 32));
    }
    if (j0 < i0 + 16) {                  // wave-uniform active check
      // ---- scores: two 16x16 tiles, each K=64 over d (2 WMMAs) ----
      v8f sc[2];
#pragma unroll
      for (int tj = 0; tj < 2; ++tj) {
        const bf16* kr = Ks[buf] + (tj * 16 + l15) * 64 + half * 16;
        v8f c = {};
        c = wmma_bf16(aq0, load_v16bf(kr), c);
        c = wmma_bf16(aq1, load_v16bf(kr + 32), c);
        sc[tj] = c;
      }
      // ---- cumulative-RPE bias + causal PAD ----
#pragma unroll
      for (int tj = 0; tj < 2; ++tj)
#pragma unroll
        for (int e = 0; e < 8; ++e) {
          int rel = (i0 + e + 8 * half) - (j0 + tj * 16 + l15);
          sc[tj][e] += (rel >= 0) ? cum_s[rel] : -30000.0f;
        }
      // ---- online softmax (rows = e + 8*half; reduce across 16-lane halves)
      float mt[8];
#pragma unroll
      for (int e = 0; e < 8; ++e) mt[e] = fmaxf(sc[0][e], sc[1][e]);
#pragma unroll
      for (int msk = 1; msk < 16; msk <<= 1)
#pragma unroll
        for (int e = 0; e < 8; ++e)
          mt[e] = fmaxf(mt[e], __shfl_xor(mt[e], msk, 32));
      float alpha[8];
#pragma unroll
      for (int e = 0; e < 8; ++e) {
        float mn = fmaxf(m_i[e], mt[e]);
        alpha[e] = exp2f((m_i[e] - mn) * 1.44269504f);
        m_i[e] = mn;
      }
#pragma unroll
      for (int tj = 0; tj < 2; ++tj)
#pragma unroll
        for (int e = 0; e < 8; ++e)
          sc[tj][e] = exp2f((sc[tj][e] - m_i[e]) * 1.44269504f);
      float rs[8];
#pragma unroll
      for (int e = 0; e < 8; ++e) rs[e] = sc[0][e] + sc[1][e];
#pragma unroll
      for (int msk = 1; msk < 16; msk <<= 1)
#pragma unroll
        for (int e = 0; e < 8; ++e) rs[e] += __shfl_xor(rs[e], msk, 32);
#pragma unroll
      for (int e = 0; e < 8; ++e) l_i[e] = l_i[e] * alpha[e] + rs[e];
#pragma unroll
      for (int dt = 0; dt < 4; ++dt)
#pragma unroll
        for (int e = 0; e < 8; ++e) o[dt][e] *= alpha[e];
      // ---- P: C-layout -> wave-private LDS -> A-layout fragment ----
#pragma unroll
      for (int tj = 0; tj < 2; ++tj)
#pragma unroll
        for (int e = 0; e < 8; ++e)
          pw[(e + 8 * half) * 48 + tj * 16 + l15] = (bf16)sc[tj][e];
      asm volatile("" ::: "memory");
      v16bf ap = load_a_frag(pw, 48, lane);
      // ---- P x V: 4 output d-tiles, K=32 over key positions ----
#pragma unroll
      for (int dt = 0; dt < 4; ++dt) {
        const bf16* vr = Vs[buf] + (dt * 16 + l15) * 32 + half * 16;
        o[dt] = wmma_bf16(ap, load_v16bf(vr), o[dt]);
      }
    }
    buf ^= 1;
  }
  // ---- normalize + store mix [b,n,s,d] bf16 (base + constant offsets) ----
#pragma unroll
  for (int e = 0; e < 8; ++e) l_i[e] = 1.0f / l_i[e];
  bf16* mb = mix + ((size_t)bn * SQ + i0 + 8 * half) * DH + l15;
#pragma unroll
  for (int dt = 0; dt < 4; ++dt)
#pragma unroll
    for (int e = 0; e < 8; ++e)
      mb[e * DH + dt * 16] = (bf16)(o[dt][e] * l_i[e]);
}

// ---------------------------------------------------------------------------
// Kernel 4: output projection. mix[b,n,s,d] (bf16) * out_w[n,d,h] -> out f32.
// A tile staged via async copies; B converted+packed; double-buffered.
// ---------------------------------------------------------------------------
__global__ __launch_bounds__(256) void out_proj_kernel(
    const bf16* __restrict__ mix, const float* __restrict__ ow,
    float* __restrict__ out) {
  __shared__ alignas(16) bf16 As[2][128 * 48];
  __shared__ alignas(16) bf16 Bs[2][128 * 48];
  int tid = threadIdx.x, wv = tid >> 5, lane = tid & 31;
  int half = lane >> 4, l15 = lane & 15;
  int r0 = blockIdx.y * 128;
  int h0 = blockIdx.x * 128;
  int wm = (wv >> 2) * 64, wn = (wv & 3) * 32;
  int arow = tid >> 1, akk = (tid & 1) * 16;
  unsigned aoff[2] = { lds_off(As[0] + arow * 48 + akk),
                       lds_off(As[1] + arow * 48 + akk) };
  int ar = r0 + arow, ab = ar >> 11, as = ar & (SQ - 1);
  const bf16* arowp = mix + ((size_t)ab * NH * SQ + as) * DH;  // + n*SQ*DH + d
  int bkp = tid >> 4, bn0 = (tid & 15) * 8;

  v8f acc[4][2] = {};
  // prologue stage (k0 = 0)
  {
    const bf16* src = arowp + akk;  // n=0, d=akk
    async_b128(aoff[0], src);
    async_b128(aoff[0] + 16, src + 8);
    const float* s0 = ow + (size_t)(2 * bkp) * HD + h0 + bn0;
    const float* s1 = s0 + HD;
#pragma unroll
    for (int j = 0; j < 8; ++j)
      *reinterpret_cast<unsigned*>(Bs[0] + (bn0 + j) * 48 + 2 * bkp) =
          pack2(s0[j], s1[j]);
  }
  int buf = 0;
  for (int k0 = 0; k0 < NH * DH; k0 += 32) {
    wait_async0();
    __syncthreads();
    if (k0 + 32 < NH * DH) {  // stage next tile into other buffer
      int kbase = k0 + 32 + akk;
      int n = kbase >> 6, d = kbase & 63;
      const bf16* src = arowp + (size_t)n * SQ * DH + d;
      async_b128(aoff[buf ^ 1], src);
      async_b128(aoff[buf ^ 1] + 16, src + 8);
      const float* s0 = ow + (size_t)(k0 + 32 + 2 * bkp) * HD + h0 + bn0;
      const float* s1 = s0 + HD;
#pragma unroll
      for (int j = 0; j < 8; ++j)
        *reinterpret_cast<unsigned*>(Bs[buf ^ 1] + (bn0 + j) * 48 + 2 * bkp) =
            pack2(s0[j], s1[j]);
    }
    v16bf af[4], bfr[2];
#pragma unroll
    for (int mi = 0; mi < 4; ++mi)
      af[mi] = load_a_frag(As[buf] + (wm + mi * 16) * 48, 48, lane);
#pragma unroll
    for (int ni = 0; ni < 2; ++ni)
      bfr[ni] = load_b_fragT(Bs[buf] + (wn + ni * 16) * 48, 48, lane);
#pragma unroll
    for (int mi = 0; mi < 4; ++mi)
#pragma unroll
      for (int ni = 0; ni < 2; ++ni)
        acc[mi][ni] = wmma_bf16(af[mi], bfr[ni], acc[mi][ni]);
    buf ^= 1;
  }
#pragma unroll
  for (int mi = 0; mi < 4; ++mi)
#pragma unroll
    for (int ni = 0; ni < 2; ++ni) {
      float* p = out + (size_t)(r0 + wm + mi * 16 + 8 * half) * HD +
                 h0 + wn + ni * 16 + l15;
#pragma unroll
      for (int e = 0; e < 8; ++e) p[e * HD] = acc[mi][ni][e];
    }
}

// ---------------------------------------------------------------------------
extern "C" void kernel_launch(void* const* d_in, const int* in_sizes, int n_in,
                              void* d_out, int out_size, void* d_ws, size_t ws_size,
                              hipStream_t stream) {
  const float* x    = (const float*)d_in[0];   // [B,S,H]
  const float* qkvw = (const float*)d_in[1];   // [H,3,N,D]
  const float* outw = (const float*)d_in[2];   // [N,D,H]
  const float* rpe  = (const float*)d_in[3];   // [N,S]

  // Workspace: cum (128 KB) + q,k,vt,mix bf16 (4 x 8 MB) = ~33.7 MB
  char* ws = (char*)d_ws;
  float* cum = (float*)ws;
  ws += (size_t)NH * SQ * sizeof(float);
  size_t qelems = (size_t)B_ * NH * SQ * DH;
  bf16* q  = (bf16*)ws; ws += qelems * sizeof(bf16);
  bf16* k  = (bf16*)ws; ws += qelems * sizeof(bf16);
  bf16* vt = (bf16*)ws; ws += qelems * sizeof(bf16);
  bf16* mix = (bf16*)ws;

  rpe_cumsum_kernel<<<NH, 32, 0, stream>>>(rpe, cum);
  qkv_proj_kernel<<<dim3(QKV_COLS / 128, (B_ * SQ) / 128), 256, 0, stream>>>(
      x, qkvw, q, k, vt);
  attn_kernel<<<dim3(SQ / 128, B_ * NH), 256, 0, stream>>>(q, k, vt, cum, mix);
  out_proj_kernel<<<dim3(HD / 128, (B_ * SQ) / 128), 256, 0, stream>>>(
      mix, outw, (float*)d_out);
}
